// LSTM_16166256902491
// MI455X (gfx1250) — matrix operand order repaired and checked by reading, not compiled, not run on hardware
//
#include <hip/hip_runtime.h>
#include <hip/hip_bf16.h>

// ---------------------------------------------------------------------------
// LSTM  B=32, T=512, D=1024, U=1024  -> out [32,512,1024] fp32
// Persistent-workgroup design: 128 WGs x 256 threads (8 waves, wave32).
// Each WG owns 8 units (=32 gate columns); its 2048x32 bf16 weight tile stays
// resident in LDS for the whole sequence. GEMM via v_wmma_f32_16x16x32_bf16,
// waves split as {M-tile}x{N-tile}x{K-half}; partials combined with ds_add_f32.
// ---------------------------------------------------------------------------

typedef __attribute__((ext_vector_type(16))) __bf16 v16bf;
typedef __attribute__((ext_vector_type(8)))  __bf16 v8bf;
typedef __attribute__((ext_vector_type(8)))  float  v8f;

#define LSTM_B   32
#define LSTM_T   512
#define LSTM_D   1024
#define LSTM_U   1024
#define LSTM_K   2048              // D + U
#define N_WG     128               // persistent workgroups
#define UPW      8                 // units per WG
#define COLS     32                // 4 gates * UPW
#define THREADS  256               // 8 waves

// Workspace layout (bytes)
#define CTR_OFF   0u
#define HBUF_OFF  256u                                   // 2 * 32*1024 bf16 = 128 KiB
#define XSEQ_OFF  (HBUF_OFF + 2u*LSTM_B*LSTM_U*2u)       // 32*512*1024 bf16 = 32 MiB
#define WT_OFF    (XSEQ_OFF + (unsigned)LSTM_B*LSTM_T*LSTM_D*2u)  // 4096*2048 bf16 = 16 MiB

// Dynamic LDS layout (bytes)
#define A_LDS_OFF 0u                         // 32*2048 bf16 = 131072
#define W_LDS_OFF 131072u                    // 32*2048 bf16 = 131072
#define Z_LDS_OFF 262144u                    // 32*32 f32    = 4096
#define SMEM_BYTES 266240u

#if __has_builtin(__builtin_amdgcn_tanhf)
__device__ __forceinline__ float fast_tanh(float x) { return __builtin_amdgcn_tanhf(x); }
#else
__device__ __forceinline__ float fast_tanh(float x) { return tanhf(x); }
#endif
__device__ __forceinline__ float fast_sig(float x) { return 0.5f * fast_tanh(0.5f * x) + 0.5f; }

// -------------------------- prep kernels -----------------------------------

__global__ void k_init(unsigned* ctr, __bf16* hbuf) {
    unsigned i = blockIdx.x * blockDim.x + threadIdx.x;
    if (i == 0) *ctr = 0u;
    unsigned n = 2u * LSTM_B * LSTM_U;
    for (unsigned k = i; k < n; k += gridDim.x * blockDim.x)
        hbuf[k] = (__bf16)0.0f;
}

__global__ void k_conv_x(const float* __restrict__ seq, __bf16* __restrict__ xseq) {
    unsigned n = (unsigned)LSTM_B * LSTM_T * LSTM_D;
    for (unsigned i = blockIdx.x * blockDim.x + threadIdx.x; i < n;
         i += gridDim.x * blockDim.x)
        xseq[i] = (__bf16)seq[i];
}

// wtile[g][c][k] (col-major per WG): c = gate*8 + local_unit, value = gateW[k][g*8+lu]
__global__ void k_conv_w(const float* __restrict__ IGw, const float* __restrict__ FGw,
                         const float* __restrict__ OGw, const float* __restrict__ CGw,
                         __bf16* __restrict__ wt) {
    unsigned n = (unsigned)N_WG * COLS * LSTM_K;     // 8.39M
    for (unsigned i = blockIdx.x * blockDim.x + threadIdx.x; i < n;
         i += gridDim.x * blockDim.x) {
        unsigned g   = i / (COLS * LSTM_K);
        unsigned rem = i % (COLS * LSTM_K);
        unsigned c   = rem / LSTM_K;
        unsigned k   = rem % LSTM_K;
        unsigned gate = c >> 3;
        unsigned unit = g * UPW + (c & 7);
        const float* W = (gate == 0) ? IGw : (gate == 1) ? FGw : (gate == 2) ? OGw : CGw;
        wt[i] = (__bf16)W[k * LSTM_U + unit];
    }
}

// -------------------------- main persistent kernel --------------------------

__global__ void __launch_bounds__(THREADS)
k_lstm(const __bf16* __restrict__ xseq,        // [B][T][D] bf16
       const __bf16* __restrict__ wt,          // [N_WG][COLS][K] bf16
       const float* __restrict__ IGb, const float* __restrict__ FGb,
       const float* __restrict__ OGb, const float* __restrict__ CGb,
       __bf16* __restrict__ hbuf,              // [2][B][U] bf16 (parity double buffer)
       unsigned* __restrict__ ctr,
       float* __restrict__ out)                // [B][T][U] fp32
{
    extern __shared__ __align__(16) char smem[];
    __bf16* Alds = (__bf16*)(smem + A_LDS_OFF);   // row-major  [32][2048]
    __bf16* Wlds = (__bf16*)(smem + W_LDS_OFF);   // col-major  [32 cols][2048]
    float*  zls  = (float*)(smem + Z_LDS_OFF);    // [32][32]

    const int tid  = threadIdx.x;
    const int wg   = blockIdx.x;
    const int u0   = wg * UPW;
    const int lane = tid & 31;
    const int wv   = tid >> 5;         // 0..7
    const int mt   = wv & 1;           // M-tile (batch 16-row half)
    const int nt   = (wv >> 1) & 1;    // N-tile (16-col half)
    const int kh   = wv >> 2;          // K-half (0: kb 0..31, 1: kb 32..63)
    const int hs   = lane >> 4;        // lane half-select
    const int arow = mt * 16 + (lane & 15);
    const int bcol = nt * 16 + (lane & 15);

    // Uniform-trip K loop: fold wave-dependent K-half into base pointers so the
    // inner loop has a compile-time trip count (no divergent-bound EXEC churn)
    // and every ds_load offset is an immediate.
    const __bf16* aptr = Alds + arow * LSTM_K + kh * 1024 + hs * 16;
    const __bf16* bptr = Wlds + bcol * LSTM_K + kh * 1024 + hs * 16;

    // per-thread recurrent cell state + biases (one (b,j) element per thread)
    const int eb = tid >> 3;           // batch 0..31
    const int ej = tid & 7;            // local unit 0..7
    float creg = 0.0f;
    const float bi = IGb[u0 + ej];
    const float bf = FGb[u0 + ej];
    const float bo = OGb[u0 + ej];
    const float bg = CGb[u0 + ej];

    // ---- one-time setup: weight tile -> LDS ----
    {
        const __bf16* wsrc = wt + (size_t)wg * COLS * LSTM_K;
        #pragma unroll 4
        for (int q = tid; q < (COLS * LSTM_K) / 8; q += THREADS)
            *(v8bf*)(Wlds + q * 8) = *(const v8bf*)(wsrc + q * 8);
    }
    __syncthreads();

    // ---- sequence loop ----
    for (int t = 0; t < LSTM_T; ++t) {
        const int par  = t & 1;
        const __bf16* hsrc = hbuf + par * (LSTM_B * LSTM_U);

        // stage A = [x_t | h_{t-1}] as bf16, row-major [32][2048]; zero z accum
        #pragma unroll 2
        for (int q = tid; q < (LSTM_B * LSTM_K) / 8; q += THREADS) {
            int lin = q * 8;
            int b   = lin >> 11;            // /2048
            int col = lin & 2047;
            v8bf v;
            if (col < LSTM_D)
                v = *(const v8bf*)(xseq + ((size_t)b * LSTM_T + t) * LSTM_D + col);
            else
                v = *(const v8bf*)(hsrc + b * LSTM_U + (col - LSTM_D));
            *(v8bf*)(Alds + lin) = v;
        }
        #pragma unroll
        for (int q = tid; q < LSTM_B * COLS; q += THREADS) zls[q] = 0.0f;
        // prefetch next step's x slice into caches (global_prefetch_b8 path)
        if (t + 1 < LSTM_T) {
            int b = tid & 31;
            __builtin_prefetch(xseq + ((size_t)b * LSTM_T + (t + 1)) * LSTM_D +
                               (tid >> 5) * 128, 0, 1);
        }
        __syncthreads();

        // ---- GEMM: z[32x32] += A[32x2048] * Wlds[2048x32]
        // wave (mt,nt,kh) does K-half kh of 16x16 tile (mt,nt); 32 iterations.
        v8f acc = {};
        #pragma unroll 8
        for (int i = 0; i < 32; ++i) {
            // A fragment (16x32 bf16): one contiguous 16-half run per lane (two 8-half loads)
            v8bf alo = *(const v8bf*)(aptr + i * 32);
            v8bf ahi = *(const v8bf*)(aptr + i * 32 + 8);
            v16bf a;
            #pragma unroll
            for (int j = 0; j < 8; ++j) { a[j] = alo[j]; a[j + 8] = ahi[j]; }
            // B fragment (32x16 bf16): one contiguous 16-half run per lane
            v16bf b = *(const v16bf*)(bptr + i * 32);
            acc = __builtin_amdgcn_wmma_f32_16x16x32_bf16(
                      false, a, false, b, (short)0, acc, false, false);
        }
        // combine K-half partials into z via LDS float atomics (ds_add_f32)
        {
            int N = nt * 16 + (lane & 15);
            #pragma unroll
            for (int r = 0; r < 8; ++r) {
                int M = mt * 16 + ((lane < 16) ? r : (8 + r));
                atomicAdd(&zls[M * COLS + N], acc[r]);
            }
        }
        __syncthreads();

        // ---- elementwise gates, c/h update, outputs (one element per thread)
        __bf16* hdst = hbuf + ((t + 1) & 1) * (LSTM_B * LSTM_U);
        {
            float ig = fast_sig(zls[eb * COLS + ej] + bi);
            float fg = fast_sig(zls[eb * COLS + 8 + ej] + bf);
            float og = fast_sig(zls[eb * COLS + 16 + ej] + bo);
            float gg = fast_tanh(zls[eb * COLS + 24 + ej] + bg);
            creg = creg * fg + gg * ig;
            float h = og * fast_tanh(creg);
            out[((size_t)eb * LSTM_T + t) * LSTM_U + u0 + ej] = h;
            hdst[eb * LSTM_U + u0 + ej] = (__bf16)h;
        }

        // ---- grid-wide step barrier (monotonic counter) ----
        __threadfence();
        __syncthreads();
        if (tid == 0) {
            atomicAdd(ctr, 1u);
            unsigned target = (unsigned)(t + 1) * gridDim.x;
            while (*(volatile unsigned*)ctr < target)
                __builtin_amdgcn_s_sleep(2);
        }
        __syncthreads();
        __threadfence();
    }
}

// ---------------------------------------------------------------------------

extern "C" void kernel_launch(void* const* d_in, const int* in_sizes, int n_in,
                              void* d_out, int out_size, void* d_ws, size_t ws_size,
                              hipStream_t stream) {
    const float* seq = (const float*)d_in[0];
    const float* IGw = (const float*)d_in[1];
    const float* IGb = (const float*)d_in[2];
    const float* FGw = (const float*)d_in[3];
    const float* FGb = (const float*)d_in[4];
    const float* OGw = (const float*)d_in[5];
    const float* OGb = (const float*)d_in[6];
    const float* CGw = (const float*)d_in[7];
    const float* CGb = (const float*)d_in[8];

    char* ws = (char*)d_ws;
    unsigned* ctr  = (unsigned*)(ws + CTR_OFF);
    __bf16*   hbuf = (__bf16*)(ws + HBUF_OFF);
    __bf16*   xseq = (__bf16*)(ws + XSEQ_OFF);
    __bf16*   wt   = (__bf16*)(ws + WT_OFF);

    k_init<<<64, 256, 0, stream>>>(ctr, hbuf);
    k_conv_x<<<2048, 256, 0, stream>>>(seq, xseq);
    k_conv_w<<<2048, 256, 0, stream>>>(IGw, FGw, OGw, CGw, wt);

    k_lstm<<<N_WG, THREADS, SMEM_BYTES, stream>>>(
        xseq, wt, IGb, FGb, OGb, CGb, hbuf, ctr, (float*)d_out);
}
